// OpenVocabularyDetectionViLD_15092515078464
// MI455X (gfx1250) — matrix operand order repaired and checked by reading, not compiled, not run on hardware
//
#include <hip/hip_runtime.h>
#include <hip/hip_bf16.h>

#define NBOX      16384
#define FEATN     512
#define NCATS     65
#define KDET      25
#define NMS_TH    0.6f
#define MAXDETS   1000
#define MINSCORE  0.9f
#define MINAREA   220.0f

typedef _Float16 v16h __attribute__((ext_vector_type(16)));
typedef _Float16 v8h  __attribute__((ext_vector_type(8)));
typedef float    v8f  __attribute__((ext_vector_type(8)));

// ---- workspace layout (byte offsets) ----
#define WS_SORTED_IDX   0          // int[16384]
#define WS_KEEP         65536      // int[16384]
#define WS_PREVALID     131072     // int[16384]
#define WS_SORTED_BOX   196608     // float[16384*4]
#define WS_IDXS         458752     // int[32]
#define WS_NV           458880     // int
#define WS_SCORES       458944     // float[25*65]

// ---------------- prep: pre-valid flags + clear keep ----------------
__global__ __launch_bounds__(256) void vild_prep_kernel(
    const float* __restrict__ roi_boxes, const float* __restrict__ roi_scores,
    const float* __restrict__ det_boxes, const float* __restrict__ image_info,
    int* __restrict__ prevalid, int* __restrict__ keep) {
  int i = blockIdx.x * blockDim.x + threadIdx.x;
  if (i >= NBOX) return;
  keep[i] = 0;
  float4 rb = reinterpret_cast<const float4*>(roi_boxes)[i];
  bool nz = !(rb.x == 0.f && rb.y == 0.f && rb.z == 0.f && rb.w == 0.f);
  float sc = roi_scores[i];
  float4 db = reinterpret_cast<const float4*>(det_boxes)[i];
  float sy = image_info[4], sx = image_info[5];
  float ry1 = db.x / sy, rx1 = db.y / sx, ry2 = db.z / sy, rx2 = db.w / sx;
  float area = (ry2 - ry1) * (rx2 - rx1);
  prevalid[i] = (nz && sc >= MINSCORE && area > MINAREA) ? 1 : 0;
}

// ---------------- bitonic sort by (-score, idx) in 128KB LDS ----------------
__global__ __launch_bounds__(1024) void vild_sort_kernel(
    const float* __restrict__ roi_scores, const float* __restrict__ det_boxes,
    int* __restrict__ sorted_idx, float* __restrict__ sorted_box) {
  extern __shared__ char smem[];
  float* ss = reinterpret_cast<float*>(smem);            // 16384 scores
  int*   si = reinterpret_cast<int*>(smem + NBOX * 4);   // 16384 indices
  int tid = threadIdx.x;
  for (int t = tid; t < NBOX; t += 1024) { ss[t] = roi_scores[t]; si[t] = t; }
  __syncthreads();
  for (int k = 2; k <= NBOX; k <<= 1) {
    for (int j = k >> 1; j > 0; j >>= 1) {
      for (int t = tid; t < NBOX; t += 1024) {
        int ixj = t ^ j;
        if (ixj > t) {
          float sa = ss[t], sb = ss[ixj];
          int   ia = si[t], ib = si[ixj];
          // total order: higher score first, tie -> smaller original index
          bool a_before_b = (sa > sb) || (sa == sb && ia < ib);
          bool up = ((t & k) == 0);
          bool doswap = up ? !a_before_b : a_before_b;
          if (doswap) { ss[t] = sb; ss[ixj] = sa; si[t] = ib; si[ixj] = ia; }
        }
      }
      __syncthreads();
    }
  }
  for (int t = tid; t < NBOX; t += 1024) {
    int oi = si[t];
    sorted_idx[t] = oi;
    reinterpret_cast<float4*>(sorted_box)[t] =
        reinterpret_cast<const float4*>(det_boxes)[oi];
  }
}

// ---------------- greedy NMS: flags in LDS, boxes in VGPRs ----------------
#define NMS_THREADS 1024
#define BPT (NBOX / NMS_THREADS)  // 16 boxes per thread
__global__ __launch_bounds__(NMS_THREADS) void vild_nms_kernel(
    const float* __restrict__ sorted_box, const int* __restrict__ sorted_idx,
    int* __restrict__ keep) {
  __shared__ unsigned char sup[NBOX];  // 16 KB suppressed flags
  int tid = threadIdx.x;
  __builtin_prefetch(sorted_box, 0, 1);  // global_prefetch_b8 hint: warm L2/WGP$
  float by1[BPT], bx1[BPT], by2[BPT], bx2[BPT], ar[BPT];
#pragma unroll
  for (int j = 0; j < BPT; ++j) {
    int p = j * NMS_THREADS + tid;
    float4 b = reinterpret_cast<const float4*>(sorted_box)[p];
    by1[j] = b.x; bx1[j] = b.y; by2[j] = b.z; bx2[j] = b.w;
    ar[j] = (b.w - b.y) * (b.z - b.x);
    sup[p] = 0;
  }
  __syncthreads();
  int cnt = 0;
  for (int i = 0; i < NBOX; ++i) {   // uniform control flow across the block
    if (cnt >= MAXDETS) break;
    if (sup[i]) continue;            // no pending LDS writes -> safe, no barrier
    float4 bi = reinterpret_cast<const float4*>(sorted_box)[i];  // uniform load
    float ai = (bi.w - bi.y) * (bi.z - bi.x);
#pragma unroll
    for (int j = 0; j < BPT; ++j) {
      int p = j * NMS_THREADS + tid;
      if (p > i) {
        float iw = fminf(bi.w, bx2[j]) - fmaxf(bi.y, bx1[j]);
        float ih = fminf(bi.z, by2[j]) - fmaxf(bi.x, by1[j]);
        iw = fmaxf(iw, 0.f); ih = fmaxf(ih, 0.f);
        float inter = iw * ih;
        float iou = inter / (ai + ar[j] - inter + 1e-12f);
        if (iou > NMS_TH) sup[p] = 1;
      }
    }
    __syncthreads();
    if (tid == 0) keep[sorted_idx[i]] = 1;
    cnt++;
  }
}

// ---------------- compaction + WMMA scores: 25x65 = A(25x512) * T(65x512)^T ----
#define SK_THREADS 320   // 10 waves -> 2 M-tiles x 5 N-tiles of 16x16
#define LDA 520          // f16 row stride in LDS (padded, 16B aligned chunks)
__global__ __launch_bounds__(SK_THREADS) void vild_score_kernel(
    const float* __restrict__ vis, const float* __restrict__ txt,
    const int* __restrict__ keep, const int* __restrict__ prevalid,
    int* __restrict__ idxs_out, int* __restrict__ nv_out,
    float* __restrict__ scores_all) {
  extern __shared__ char smem[];
  _Float16* Abuf = reinterpret_cast<_Float16*>(smem);                  // 32 rows
  _Float16* Bbuf = reinterpret_cast<_Float16*>(smem + 32 * LDA * 2);   // 80 rows
  __shared__ int s_idx[KDET];
  int tid = threadIdx.x;
  if (tid == 0) {  // ordered compaction of first K valid indices (reference: nonzero)
    int cnt = 0;
    for (int i = 0; i < NBOX; ++i) {
      if (keep[i] && prevalid[i]) {
        if (cnt < KDET) s_idx[cnt] = i;
        cnt++;
      }
    }
    for (int r = cnt; r < KDET; ++r) s_idx[r] = 0;  // fill_value=0
    nv_out[0] = cnt < KDET ? cnt : KDET;
    for (int r = 0; r < KDET; ++r) idxs_out[r] = s_idx[r];
  }
  __syncthreads();
  for (int t = tid; t < 32 * FEATN; t += SK_THREADS) {
    int r = t >> 9, k = t & 511;
    float v = (r < KDET) ? vis[s_idx[r] * FEATN + k] : 0.f;
    Abuf[r * LDA + k] = (_Float16)v;
  }
  for (int t = tid; t < 80 * FEATN; t += SK_THREADS) {
    int r = t / FEATN, k = t % FEATN;
    float v = (r < NCATS) ? txt[r * FEATN + k] : 0.f;
    Bbuf[r * LDA + k] = (_Float16)v;
  }
  __syncthreads();
  int wid = tid >> 5, lane = tid & 31;
  int mt = wid / 5, nt = wid % 5;
  int hi = lane >> 4;                 // lane half select
  int mrow = mt * 16 + (lane & 15);   // A: M = lane&15
  int nrow = nt * 16 + (lane & 15);   // B: N = lane&15
  v8f acc = {};
#pragma unroll
  for (int kk = 0; kk < FEATN / 32; ++kk) {
    // A 16x32 f16 layout: lanes<16 hold K {0..7,16..23}, lanes>=16 hold {8..15,24..31}
    int ka = kk * 32 + hi * 8;
    v8h a0 = *reinterpret_cast<const v8h*>(Abuf + mrow * LDA + ka);
    v8h a1 = *reinterpret_cast<const v8h*>(Abuf + mrow * LDA + ka + 16);
    // B 32x16 f16 layout: lanes<16 hold K {0..7,16..23}, lanes>=16 hold {8..15,24..31}
    int kb = kk * 32 + hi * 8;
    v8h b0 = *reinterpret_cast<const v8h*>(Bbuf + nrow * LDA + kb);
    v8h b1 = *reinterpret_cast<const v8h*>(Bbuf + nrow * LDA + kb + 16);
    v16h av, bv;
#pragma unroll
    for (int j = 0; j < 8; ++j) {
      av[j] = a0[j]; av[j + 8] = a1[j];
      bv[j] = b0[j]; bv[j + 8] = b1[j];
    }
    acc = __builtin_amdgcn_wmma_f32_16x16x32_f16(
        false, av, false, bv, (short)0, acc, false, false);
  }
  // C/D f32 16x16: VGPR g -> M = g + 8*(lane>=16), N = lane&15
#pragma unroll
  for (int g = 0; g < 8; ++g) {
    int m = mt * 16 + g + hi * 8;
    int n = nt * 16 + (lane & 15);
    if (m < KDET && n < NCATS) scores_all[m * NCATS + n] = acc[g];
  }
}

// ---------------- final: argmax, stable descending rank, masked outputs ------
__global__ __launch_bounds__(32) void vild_final_kernel(
    const float* __restrict__ scores_all, const int* __restrict__ idxs,
    const int* __restrict__ nv_p, const float* __restrict__ det_boxes,
    const float* __restrict__ image_info, float* __restrict__ out) {
  __shared__ float s_key[KDET];
  __shared__ int s_fg[KDET];
  __shared__ int s_ord[KDET];
  int r = threadIdx.x;
  int nv = nv_p[0];
  if (r < KDET) {
    float best = -3.0e38f; int bi = 0;
    for (int c = 0; c < NCATS; ++c) {
      float s = scores_all[r * NCATS + c];
      if (s > best) { best = s; bi = c; }   // first-occurrence argmax
    }
    int fg = (r < nv) && (bi != 0);
    s_fg[r] = fg;
    s_key[r] = fg ? best : -3.0e38f;        // -inf sentinel for sort key
  }
  __syncthreads();
  if (r < KDET) {  // stable descending rank (matches stable argsort(-key))
    float kr = s_key[r];
    int rank = 0;
    for (int q = 0; q < KDET; ++q) {
      float kq = s_key[q];
      if (kq > kr || (kq == kr && q < r)) rank++;
    }
    s_ord[rank] = r;
  }
  __syncthreads();
  if (r < KDET) {
    int src = s_ord[r];
    bool m = s_fg[src] != 0;
    for (int c = 0; c < 64; ++c)
      out[r * 64 + c] = m ? scores_all[src * NCATS + 1 + c] : 0.f;
    float sy = image_info[4], sx = image_info[5];
    int oi = idxs[src];
    float y1 = det_boxes[oi * 4 + 0] / sy, x1 = det_boxes[oi * 4 + 1] / sx;
    float y2 = det_boxes[oi * 4 + 2] / sy, x2 = det_boxes[oi * 4 + 3] / sx;
    float* bb = out + KDET * 64 + r * 4;
    bb[0] = m ? x1 : 0.f; bb[1] = m ? y1 : 0.f;
    bb[2] = m ? x2 : 0.f; bb[3] = m ? y2 : 0.f;
    out[KDET * 64 + KDET * 4 + r] = m ? 1.f : 0.f;
  }
}

extern "C" void kernel_launch(void* const* d_in, const int* in_sizes, int n_in,
                              void* d_out, int out_size, void* d_ws, size_t ws_size,
                              hipStream_t stream) {
  (void)in_sizes; (void)n_in; (void)out_size; (void)ws_size;
  const float* roi_boxes  = (const float*)d_in[0];
  const float* roi_scores = (const float*)d_in[1];
  const float* det_boxes  = (const float*)d_in[2];
  // d_in[3] detection_masks: unused by the reference outputs
  const float* vis        = (const float*)d_in[4];
  const float* image_info = (const float*)d_in[5];
  const float* txt        = (const float*)d_in[6];

  char* ws = (char*)d_ws;
  int*   sorted_idx = (int*)(ws + WS_SORTED_IDX);
  int*   keep       = (int*)(ws + WS_KEEP);
  int*   prevalid   = (int*)(ws + WS_PREVALID);
  float* sorted_box = (float*)(ws + WS_SORTED_BOX);
  int*   idxs       = (int*)(ws + WS_IDXS);
  int*   nv         = (int*)(ws + WS_NV);
  float* scores_all = (float*)(ws + WS_SCORES);
  float* out        = (float*)d_out;

  vild_prep_kernel<<<NBOX / 256, 256, 0, stream>>>(
      roi_boxes, roi_scores, det_boxes, image_info, prevalid, keep);
  vild_sort_kernel<<<1, 1024, NBOX * 8, stream>>>(  // 128 KB LDS (of 320 KB/WGP)
      roi_scores, det_boxes, sorted_idx, sorted_box);
  vild_nms_kernel<<<1, NMS_THREADS, 0, stream>>>(sorted_box, sorted_idx, keep);
  size_t smem = (size_t)(32 * LDA + 80 * LDA) * sizeof(_Float16);  // ~114 KB
  vild_score_kernel<<<1, SK_THREADS, smem, stream>>>(
      vis, txt, keep, prevalid, idxs, nv, scores_all);
  vild_final_kernel<<<1, 32, 0, stream>>>(
      scores_all, idxs, nv, det_boxes, image_info, out);
}
